// PNN_48292612276684
// MI455X (gfx1250) — matrix operand order repaired and checked by reading, not compiled, not run on hardware
//
#include <hip/hip_runtime.h>
#include <cstdint>
#include <cstddef>

// ---------------------------------------------------------------------------
// MI455X (gfx1250) PNN forward.
// 206 GFLOP of GEMM vs ~200MB weight traffic -> compute bound -> bf16 WMMA
// (v_wmma_f32_16x16x32_bf16). bf16 weights (<=32MB/layer) are L2-resident
// (192MB), so per-M-tile re-reads never hit HBM.
// GEMM loop: double-buffered LDS, global loads for tile i+1 issued before
// computing tile i, one barrier per K-step; 64x64 per wave -> 1.0
// ds_load_b128 per WMMA.
// ---------------------------------------------------------------------------

typedef __bf16 bf16;
typedef __attribute__((ext_vector_type(16))) __bf16 bf16x16;
typedef __attribute__((ext_vector_type(8)))  __bf16 bf16x8;
typedef __attribute__((ext_vector_type(4)))  __bf16 bf16x4;
typedef __attribute__((ext_vector_type(8)))  float  f32x8;

#define BATCH 2048

// ---------------- fp32 -> bf16 elementwise (input activations) -------------
__global__ __launch_bounds__(256) void cvt_x_kernel(const float* __restrict__ x,
                                                    bf16* __restrict__ out, int n) {
    int i = (blockIdx.x * blockDim.x + threadIdx.x) * 4;
    if (i < n) {
        float4 v = *(const float4*)(x + i);
        bf16x4 o; o.x = (bf16)v.x; o.y = (bf16)v.y; o.z = (bf16)v.z; o.w = (bf16)v.w;
        *(bf16x4*)(out + i) = o;
    }
}

// -------- W[K][N] fp32 -> Wt[N][K] bf16 (K contiguous for WMMA B frags) ----
__global__ __launch_bounds__(256) void cvt_wt_kernel(const float* __restrict__ W,
                                                     bf16* __restrict__ Wt,
                                                     int K, int N) {
    __shared__ bf16 tile[32][33];
    int kb = blockIdx.y * 32, nb = blockIdx.x * 32;
#pragma unroll
    for (int i = 0; i < 4; ++i) {
        int k = kb + threadIdx.y + i * 8;
        tile[threadIdx.y + i * 8][threadIdx.x] = (bf16)W[(size_t)k * N + nb + threadIdx.x];
    }
    __syncthreads();
#pragma unroll
    for (int i = 0; i < 4; ++i) {
        int n = nb + threadIdx.y + i * 8;
        Wt[(size_t)n * K + kb + threadIdx.x] = tile[threadIdx.x][threadIdx.y + i * 8];
    }
}

// deterministic per-element "bernoulli" uniform (reference RNG unreproducible)
__device__ __forceinline__ float uniform_hash(uint32_t idx, uint32_t layer) {
    uint32_t h = idx ^ (layer * 0x9E3779B9u + 0x85EBCA6Bu);
    h ^= h >> 16; h *= 0x7FEB352Du;
    h ^= h >> 15; h *= 0x846CA68Bu;
    h ^= h >> 16;
    return (float)h * 2.3283064365386963e-10f; // [0,1)
}

// ---------------------------------------------------------------------------
// Fused layer: out = pnn(sigmoid(A @ Wt^T + b))
// A:[M=2048][K] bf16 row-major, Wt:[N][K] bf16 (K contiguous).
// Block tile 128x128, 128 threads = 4 wave32 in a 2(M) x 2(N) grid,
// each wave owns 64x64 = 4x4 WMMA 16x16 fragments. K-step = 32.
// Double-buffered LDS; one barrier per K-step.
// ---------------------------------------------------------------------------
#define LDS_STRIDE 48  // 32 valid + pad; 96B row (16B aligned) for b128 LDS ops

__global__ __launch_bounds__(128) void pnn_layer_kernel(
    const bf16* __restrict__ A, const bf16* __restrict__ Wt,
    const float* __restrict__ bias,
    bf16* __restrict__ outBf, float* __restrict__ outF,
    int K, int N, int layer) {

    __shared__ bf16 sA[2][128][LDS_STRIDE];
    __shared__ bf16 sW[2][128][LDS_STRIDE];

    const int tileM = blockIdx.y * 128;
    const int tileN = blockIdx.x * 128;

    const int t     = threadIdx.x;
    const int lane  = t & 31;
    const int wave  = t >> 5;          // 0..3
    const int waveM = wave >> 1;       // 0..1  -> 64-row slab
    const int waveN = wave & 1;        // 0..1  -> 64-col slab
    const int l16   = lane & 15;
    const int hgrp  = lane >> 4;       // 0/1: lane half-group (ISA frag layout)

    f32x8 acc[4][4];
#pragma unroll
    for (int mi = 0; mi < 4; ++mi)
#pragma unroll
        for (int ni = 0; ni < 4; ++ni) acc[mi][ni] = (f32x8)0.0f;

    // global->LDS staging: 128 threads, each covers 4 rows x 8 bf16 per matrix
    const int r0 = t >> 2;             // 0..31
    const int q8 = (t & 3) * 8;

    const bf16* gA = A  + (size_t)(tileM + r0) * K + q8;
    const bf16* gW = Wt + (size_t)(tileN + r0) * K + q8;

    bf16x8 rA[4], rW[4];

    const int nk = K >> 5;             // K-steps of 32

    // stage tile 0
#pragma unroll
    for (int rr = 0; rr < 4; ++rr) {
        rA[rr] = *(const bf16x8*)(gA + (size_t)(rr * 32) * K);
        rW[rr] = *(const bf16x8*)(gW + (size_t)(rr * 32) * K);
    }
#pragma unroll
    for (int rr = 0; rr < 4; ++rr) {
        *(bf16x8*)&sA[0][r0 + rr * 32][q8] = rA[rr];
        *(bf16x8*)&sW[0][r0 + rr * 32][q8] = rW[rr];
    }

    for (int i = 0; i < nk; ++i) {
        const int cur = i & 1;
        const bool more = (i + 1) < nk;

        // issue next tile's global loads before computing this tile
        if (more) {
            const int k1 = (i + 1) << 5;
#pragma unroll
            for (int rr = 0; rr < 4; ++rr) {
                rA[rr] = *(const bf16x8*)(gA + (size_t)(rr * 32) * K + k1);
                rW[rr] = *(const bf16x8*)(gW + (size_t)(rr * 32) * K + k1);
            }
        }

        __syncthreads();   // buf[cur] filled; buf[cur^1] free to overwrite

        // A fragments: 16x32 bf16, lane row = l16, K = hgrp*8 + [0..7] and +16
        bf16x16 afrag[4];
#pragma unroll
        for (int mi = 0; mi < 4; ++mi) {
            const bf16* base = &sA[cur][waveM * 64 + mi * 16 + l16][hgrp * 8];
            union { bf16x16 v; bf16x8 h[2]; } u;
            u.h[0] = *(const bf16x8*)base;
            u.h[1] = *(const bf16x8*)(base + 16);
            afrag[mi] = u.v;
        }
        // B fragments: 32x16 bf16, lane col = l16, K = hgrp*8 + [0..7] and +16
        bf16x16 bfrag[4];
#pragma unroll
        for (int ni = 0; ni < 4; ++ni) {
            const bf16* base = &sW[cur][waveN * 64 + ni * 16 + l16][hgrp * 8];
            union { bf16x16 v; bf16x8 h[2]; } u;
            u.h[0] = *(const bf16x8*)base;
            u.h[1] = *(const bf16x8*)(base + 16);
            bfrag[ni] = u.v;
        }

#pragma unroll
        for (int mi = 0; mi < 4; ++mi)
#pragma unroll
            for (int ni = 0; ni < 4; ++ni)
                acc[mi][ni] = __builtin_amdgcn_wmma_f32_16x16x32_bf16(
                    false, afrag[mi], false, bfrag[ni],
                    (short)0, acc[mi][ni], false, false);

        // store next tile into the other buffer (read again only after the
        // barrier at the top of the next iteration)
        if (more) {
            const int nxt = cur ^ 1;
#pragma unroll
            for (int rr = 0; rr < 4; ++rr) {
                *(bf16x8*)&sA[nxt][r0 + rr * 32][q8] = rA[rr];
                *(bf16x8*)&sW[nxt][r0 + rr * 32][q8] = rW[rr];
            }
        }
    }

    // ---------------- fused epilogue: bias + sigmoid + bernoulli mix --------
#pragma unroll
    for (int ni = 0; ni < 4; ++ni) {
        const int col = tileN + waveN * 64 + ni * 16 + l16;
        const float bv = bias[col];
#pragma unroll
        for (int mi = 0; mi < 4; ++mi) {
            const int rowbase = tileM + waveM * 64 + mi * 16 + hgrp * 8;
#pragma unroll
            for (int r = 0; r < 8; ++r) {
                const int row = rowbase + r;
                const float v = acc[mi][ni][r] + bv;
                const float a = 1.0f / (1.0f + __expf(-v));
                const float u = uniform_hash((uint32_t)(row * N + col), (uint32_t)layer);
                const float res = (u < a) ? a : 1.0f; // p*a + (1-p)
                if (outF) outF[(size_t)row * N + col] = res;
                else      outBf[(size_t)row * N + col] = (bf16)res;
            }
        }
    }
}

// ---------------------------------------------------------------------------
extern "C" void kernel_launch(void* const* d_in, const int* in_sizes, int n_in,
                              void* d_out, int out_size, void* d_ws, size_t ws_size,
                              hipStream_t stream) {
    const float* x  = (const float*)d_in[0];
    const float* W1 = (const float*)d_in[1]; const float* b1 = (const float*)d_in[2];
    const float* W2 = (const float*)d_in[3]; const float* b2 = (const float*)d_in[4];
    const float* W3 = (const float*)d_in[5]; const float* b3 = (const float*)d_in[6];
    const float* W4 = (const float*)d_in[7]; const float* b4 = (const float*)d_in[8];
    float* out = (float*)d_out;

    // ws layout: Wt scratch (32MB, reused per layer; stream order serializes)
    // + two bf16 activation ping-pong buffers (16MB each). Total 64 MiB.
    char* ws   = (char*)d_ws;
    bf16* Wt   = (bf16*)ws;
    bf16* actA = (bf16*)(ws + ((size_t)32 << 20));
    bf16* actB = (bf16*)(ws + ((size_t)48 << 20));

    const dim3 tblk(32, 8);

    // x (fp32) -> bf16
    {
        int n = BATCH * 2048;
        cvt_x_kernel<<<n / 1024, 256, 0, stream>>>(x, actA, n);
    }
    // layer 1: 2048 -> 4096
    cvt_wt_kernel<<<dim3(4096 / 32, 2048 / 32), tblk, 0, stream>>>(W1, Wt, 2048, 4096);
    pnn_layer_kernel<<<dim3(4096 / 128, BATCH / 128), 128, 0, stream>>>(
        actA, Wt, b1, actB, nullptr, 2048, 4096, 0);
    // layer 2: 4096 -> 4096
    cvt_wt_kernel<<<dim3(4096 / 32, 4096 / 32), tblk, 0, stream>>>(W2, Wt, 4096, 4096);
    pnn_layer_kernel<<<dim3(4096 / 128, BATCH / 128), 128, 0, stream>>>(
        actB, Wt, b2, actA, nullptr, 4096, 4096, 1);
    // layer 3: 4096 -> 4096
    cvt_wt_kernel<<<dim3(4096 / 32, 4096 / 32), tblk, 0, stream>>>(W3, Wt, 4096, 4096);
    pnn_layer_kernel<<<dim3(4096 / 128, BATCH / 128), 128, 0, stream>>>(
        actA, Wt, b3, actB, nullptr, 4096, 4096, 2);
    // layer 4: 4096 -> 2048, fp32 straight to d_out
    cvt_wt_kernel<<<dim3(2048 / 32, 4096 / 32), tblk, 0, stream>>>(W4, Wt, 4096, 2048);
    pnn_layer_kernel<<<dim3(2048 / 128, BATCH / 128), 128, 0, stream>>>(
        actB, Wt, b4, nullptr, out, 4096, 2048, 3);
}
